// RWKVMemoryLM_45354854646019
// MI455X (gfx1250) — matrix operand order repaired and checked by reading, not compiled, not run on hardware
//
#include <hip/hip_runtime.h>
#include <hip/hip_bf16.h>
#include <math.h>

// ---------------------------------------------------------------------------
// RWKV-memory-LM forward for MI455X (gfx1250, wave32, WMMA).
// All GEMMs run as bf16 WMMA (v_wmma_f32_16x16x32_bf16) with f32 accum:
// weights fit in the 192MB L2, so GEMMs are compute-bound -> matrix cores.
// GEMM: 32x64 tile per wave (8 WMMA / K-step), register double-buffered so
// global_load_b128 overlaps v_wmma issue instead of s_wait_loadcnt 0 stalls.
// Elementwise / LN / conv / top-k / GRU stay fp32 (bandwidth-bound).
// ---------------------------------------------------------------------------

typedef __bf16 bf16;
typedef __attribute__((ext_vector_type(16))) __bf16 v16bf;
typedef __attribute__((ext_vector_type(8)))  float  v8f;
typedef __attribute__((ext_vector_type(4)))  float  f32x4;

#define BB    2
#define TT    2048
#define DD    512
#define NLAY  4
#define SS    32
#define TOPKK 4
#define VV    32000
#define LKK   64
#define MT    (BB*TT)      // 4096 rows
#define RESF  0.5f

union Frag { v16bf v; f32x4 q[2]; };

__device__ __forceinline__ float sigmf(float x) { return 1.0f / (1.0f + __expf(-x)); }

// ---------------------------------------------------------------------------
// Wave-tiled WMMA GEMM:  out[M,N] = epi( A[M,K] @ W[N,K]^T + bias )
// Block = 128 threads = 4 waves stacked in M; each wave owns 32(M) x 64(N).
// Requires: M % 32 == 0 (or wave-guard), N % 64 == 0, K % 64 == 0.
// A fragment (16x32 bf16): lane<16 holds K {s0..s0+15} contiguous,
// s0=0; lane>=16 same with s0=16. B fragment (32x16): lane holds 16 contiguous
// K of row N (row-major N,K weight => contiguous 32B load).
// ---------------------------------------------------------------------------
enum { EPI_NONE = 0, EPI_SIG = 1, EPI_SILU = 2, EPI_RES = 3 };

template <int EPI>
__global__ void __launch_bounds__(128)
k_gemm(const bf16* __restrict__ A, const bf16* __restrict__ W,
       const float* __restrict__ bias, float* Cf, bf16* Ch,
       const float* resid, int M, int N, int K)
{
    const int lane = threadIdx.x & 31;
    const int wid  = threadIdx.x >> 5;
    const int n0 = blockIdx.x * 64;
    const int m0 = blockIdx.y * 128 + wid * 32;
    if (m0 >= M) return;                       // whole-wave guard

    v8f acc[8];
#pragma unroll
    for (int i = 0; i < 8; ++i) acc[i] = v8f{};

    const int s0 = (lane < 16) ? 0 : 16;
    const bf16* arow0 = A + (size_t)(m0 + (lane & 15)) * K;
    const bf16* arow1 = A + (size_t)(m0 + 16 + (lane & 15)) * K;
    const int kb = (lane >> 4) << 4;           // 0 or 16
    const bf16* brow[4];
#pragma unroll
    for (int j = 0; j < 4; ++j)
        brow[j] = W + (size_t)(n0 + 16 * j + (lane & 15)) * K + kb;

    Frag a0c, a1c, bc[4], a0n, a1n, bn[4];

    auto lda = [&](Frag& a0, Frag& a1, int k) {
        a0.q[0] = *(const f32x4*)(arow0 + k + s0);
        a0.q[1] = *(const f32x4*)(arow0 + k + s0 + 8);
        a1.q[0] = *(const f32x4*)(arow1 + k + s0);
        a1.q[1] = *(const f32x4*)(arow1 + k + s0 + 8);
    };
    auto ldb = [&](Frag* b, int k) {
#pragma unroll
        for (int j = 0; j < 4; ++j) {
            b[j].q[0] = *(const f32x4*)(brow[j] + k);
            b[j].q[1] = *(const f32x4*)(brow[j] + k + 8);
        }
    };
    auto mma = [&](Frag& a0, Frag& a1, Frag* b) {
#pragma unroll
        for (int j = 0; j < 4; ++j) {
            acc[j] = __builtin_amdgcn_wmma_f32_16x16x32_bf16(
                false, a0.v, false, b[j].v, (short)0, acc[j], false, false);
            acc[4 + j] = __builtin_amdgcn_wmma_f32_16x16x32_bf16(
                false, a1.v, false, b[j].v, (short)0, acc[4 + j], false, false);
        }
    };

    // software pipeline over K (K % 64 == 0): loads for step k+32 are in
    // flight while the 8 WMMAs for step k execute.
    lda(a0c, a1c, 0);
    ldb(bc, 0);
    for (int k = 0; k < K; k += 64) {
        lda(a0n, a1n, k + 32);
        ldb(bn, k + 32);
        mma(a0c, a1c, bc);
        if (k + 64 < K) {
            lda(a0c, a1c, k + 64);
            ldb(bc, k + 64);
        }
        mma(a0n, a1n, bn);
    }

    const int rb = m0 + ((lane >> 4) << 3);
    const int cb = n0 + (lane & 15);
#pragma unroll
    for (int half = 0; half < 2; ++half) {
#pragma unroll
        for (int j = 0; j < 4; ++j) {
            const int col = cb + 16 * j;
            const float bv = bias ? bias[col] : 0.0f;
#pragma unroll
            for (int r = 0; r < 8; ++r) {
                const int row = rb + 16 * half + r;
                const size_t o = (size_t)row * N + col;
                float val = acc[half * 4 + j][r] + bv;
                if (EPI == EPI_SIG)  val = sigmf(val);
                if (EPI == EPI_SILU) val = val * sigmf(val);
                if (EPI == EPI_RES)  val = resid[o] + RESF * val;
                if (Cf) Cf[o] = val;
                if (Ch) Ch[o] = (bf16)val;
            }
        }
    }
}

// ---------------------------------------------------------------------------
// Small helper kernels
// ---------------------------------------------------------------------------
__global__ void k_cvt(const float* __restrict__ s, bf16* __restrict__ d, int n)
{
    int i = blockIdx.x * 256 + threadIdx.x;
    if (i < n) d[i] = (bf16)s[i];
}

__global__ void k_embed(const int* __restrict__ tok, const float* __restrict__ emb,
                        const float* __restrict__ pos, float* __restrict__ out)
{
    size_t i = (size_t)blockIdx.x * 256 + threadIdx.x;
    if (i >= (size_t)MT * DD) return;
    int d = (int)(i % DD);
    size_t row = i / DD;
    int t = (int)(row % TT);
    int id = tok[row];
    out[i] = emb[(size_t)id * DD + d] + pos[(size_t)t * DD + d];
}

// row-wise LayerNorm, optional f32 + bf16 outputs
__global__ void k_ln(const float* __restrict__ in, const float* __restrict__ g,
                     const float* __restrict__ b, float* outf, bf16* outh)
{
    const int row = blockIdx.x;
    const float* r = in + (size_t)row * DD;
    __shared__ float s1[256], s2[256];
    float a0 = 0.f, a1 = 0.f;
    for (int d = threadIdx.x; d < DD; d += 256) { float v = r[d]; a0 += v; a1 += v * v; }
    s1[threadIdx.x] = a0; s2[threadIdx.x] = a1;
    __syncthreads();
    for (int o = 128; o; o >>= 1) {
        if (threadIdx.x < o) { s1[threadIdx.x] += s1[threadIdx.x + o]; s2[threadIdx.x] += s2[threadIdx.x + o]; }
        __syncthreads();
    }
    const float m   = s1[0] / DD;
    const float var = s2[0] / DD - m * m;
    const float inv = rsqrtf(var + 1e-5f);
    for (int d = threadIdx.x; d < DD; d += 256) {
        float v = (r[d] - m) * inv * g[d] + b[d];
        size_t o = (size_t)row * DD + d;
        if (outf) outf[o] = v;
        if (outh) outh[o] = (bf16)v;
    }
}

// time/channel shift mix -> bf16 GEMM input
__global__ void k_mix(const float* __restrict__ xln, const float* __restrict__ mix,
                      bf16* __restrict__ out)
{
    size_t i = (size_t)blockIdx.x * 256 + threadIdx.x;
    if (i >= (size_t)MT * DD) return;
    int d = (int)(i % DD);
    size_t row = i / DD;
    int t = (int)(row % TT);
    float prev = (t > 0) ? xln[i - DD] : 0.0f;
    float tm = mix[d];
    out[i] = (bf16)(xln[i] * (1.0f - tm) + prev * tm);
}

__global__ void k_rv(float* __restrict__ v, const float* __restrict__ r)
{
    size_t i = (size_t)blockIdx.x * 256 + threadIdx.x;
    if (i < (size_t)MT * DD) v[i] *= r[i];
}

// causal exponential-decay conv:  y[t] = sum_{j<L} decay^j * rv[t-j]
__global__ void k_decay(const float* __restrict__ rv, const float* __restrict__ dl,
                        bf16* __restrict__ y)
{
    size_t i = (size_t)blockIdx.x * 256 + threadIdx.x;
    if (i >= (size_t)MT * DD) return;
    int d = (int)(i % DD);
    size_t row = i / DD;
    int t = (int)(row % TT);
    const float w = sigmf(dl[d]);
    const int L = (t + 1 < LKK) ? (t + 1) : LKK;
    float acc = 0.f, p = 1.f;
    for (int j = 0; j < L; ++j) {
        acc += p * rv[(row - j) * DD + d];
        p *= w;
    }
    y[i] = (bf16)acc;
}

// depthwise dilated conv taps {t-4, t-2, t} + bias -> bf16
__global__ void k_dw(const float* __restrict__ xl, const float* __restrict__ w,
                     const float* __restrict__ b, bf16* __restrict__ out)
{
    size_t i = (size_t)blockIdx.x * 256 + threadIdx.x;
    if (i >= (size_t)MT * DD) return;
    int d = (int)(i % DD);
    size_t row = i / DD;
    int t = (int)(row % TT);
    float x0 = (t >= 4) ? xl[i - 4 * DD] : 0.f;
    float x1 = (t >= 2) ? xl[i - 2 * DD] : 0.f;
    out[i] = (bf16)(w[d * 3] * x0 + w[d * 3 + 1] * x1 + w[d * 3 + 2] * xl[i] + b[d]);
}

// one wave per (b,t): scores over 32 slots, top-4 masked softmax, ctx = attn@V
__global__ void k_attn(const float* __restrict__ q, const float* __restrict__ kb,
                       const float* __restrict__ vb, bf16* __restrict__ ctx)
{
    const int row = blockIdx.x;           // b*T + t
    const int s = threadIdx.x;            // slot = lane, wave32
    const float* qr = q + (size_t)row * DD;
    const float* kr = kb + (size_t)s * DD;
    float sc = 0.f;
    for (int d = 0; d < DD; ++d) sc += qr[d] * kr[d];
    sc *= 0.044194173824159216f;          // 1/sqrt(512)

    // top-4 selection with lane tie-break
    float cur = sc;
    bool sel = false;
    for (int it = 0; it < TOPKK; ++it) {
        float m = cur;
        for (int off = 16; off; off >>= 1) m = fmaxf(m, __shfl_xor(m, off, 32));
        int cand = (cur == m) ? s : 64;
        for (int off = 16; off; off >>= 1) cand = min(cand, __shfl_xor(cand, off, 32));
        if (s == cand) { sel = true; cur = -3.0e38f; }
    }
    float mx = sel ? sc : -3.0e38f;
    for (int off = 16; off; off >>= 1) mx = fmaxf(mx, __shfl_xor(mx, off, 32));
    float e = sel ? __expf(sc - mx) : 0.f;
    float den = e;
    for (int off = 16; off; off >>= 1) den += __shfl_xor(den, off, 32);
    const float attn = e / den;

    __shared__ float asml[SS];
    asml[s] = attn;
    __syncthreads();

    bf16* cr = ctx + (size_t)row * DD;
    for (int d = s; d < DD; d += 32) {
        float acc = 0.f;
        for (int t2 = 0; t2 < SS; ++t2) acc += asml[t2] * vb[(size_t)t2 * DD + d];
        cr[d] = (bf16)acc;
    }
}

// memory gate: x += sigmoid(mg_w . [x,ctx] + mg_b + rel[t]) * ctx
__global__ void k_gate(float* x, const float* __restrict__ ctx,
                       const float* __restrict__ mgw, const float* __restrict__ mgb,
                       const float* __restrict__ rel)
{
    const int row = blockIdx.x;
    const int t = row % TT;
    const float* xr = x + (size_t)row * DD;
    const float* cr = ctx + (size_t)row * DD;
    __shared__ float red[256];
    float p = 0.f;
    for (int d = threadIdx.x; d < DD; d += 256)
        p += mgw[d] * xr[d] + mgw[DD + d] * cr[d];
    red[threadIdx.x] = p;
    __syncthreads();
    for (int o = 128; o; o >>= 1) {
        if (threadIdx.x < o) red[threadIdx.x] += red[threadIdx.x + o];
        __syncthreads();
    }
    const float g = sigmf(red[0] + mgb[0] + rel[t]);
    for (int d = threadIdx.x; d < DD; d += 256)
        x[(size_t)row * DD + d] = xr[d] + g * cr[d];
}

// mean over middle axis: in (G,R,D) -> out (G,D)
__global__ void k_mean(const float* __restrict__ in, float* __restrict__ out,
                       int G, int R, int D)
{
    int i = blockIdx.x * 256 + threadIdx.x;
    if (i >= G * D) return;
    int g = i / D, d = i % D;
    float a = 0.f;
    for (int r = 0; r < R; ++r) a += in[((size_t)g * R + r) * D + d];
    out[i] = a / (float)R;
}

__global__ void k_gscal(const float* __restrict__ summ, const float* __restrict__ mm,
                        const float* __restrict__ gw, const float* __restrict__ gb,
                        float* __restrict__ g)
{
    const int b = blockIdx.x;
    __shared__ float red[256];
    float p = 0.f;
    for (int d = threadIdx.x; d < DD; d += 256)
        p += gw[d] * summ[b * DD + d] + gw[DD + d] * mm[d];
    red[threadIdx.x] = p;
    __syncthreads();
    for (int o = 128; o; o >>= 1) {
        if (threadIdx.x < o) red[threadIdx.x] += red[threadIdx.x + o];
        __syncthreads();
    }
    if (threadIdx.x == 0) g[b] = sigmf(red[0] + gb[0]);
}

// GRU projections: gi[b,3D] = wih@summary[b] + bih ; gh[s,3D] = whh@slots[s] + bhh
__global__ void k_gruproj(const float* __restrict__ summ, const float* __restrict__ slots,
                          const float* __restrict__ wih, const float* __restrict__ bih,
                          const float* __restrict__ whh, const float* __restrict__ bhh,
                          float* __restrict__ gi, float* __restrict__ gh)
{
    const int NI = BB * 3 * DD;
    const int NH = SS * 3 * DD;
    int i = blockIdx.x * 256 + threadIdx.x;
    if (i < NI) {
        int b = i / (3 * DD), j = i % (3 * DD);
        float a = bih[j];
        const float* wr = wih + (size_t)j * DD;
        const float* xr = summ + b * DD;
        for (int d = 0; d < DD; ++d) a += wr[d] * xr[d];
        gi[i] = a;
    } else if (i < NI + NH) {
        int ii = i - NI;
        int s = ii / (3 * DD), j = ii % (3 * DD);
        float a = bhh[j];
        const float* wr = whh + (size_t)j * DD;
        const float* xr = slots + s * DD;
        for (int d = 0; d < DD; ++d) a += wr[d] * xr[d];
        gh[ii] = a;
    }
}

__global__ void k_gruupd(const float* __restrict__ gi, const float* __restrict__ gh,
                         const float* __restrict__ slots, const float* __restrict__ g,
                         float* __restrict__ out)
{
    int i = blockIdx.x * 256 + threadIdx.x;
    if (i >= BB * SS * DD) return;
    int d = i % DD;
    int s = (i / DD) % SS;
    int b = i / (SS * DD);
    float ir = gi[b * 3 * DD + d], iz = gi[b * 3 * DD + DD + d], in = gi[b * 3 * DD + 2 * DD + d];
    float hr = gh[s * 3 * DD + d], hz = gh[s * 3 * DD + DD + d], hn = gh[s * 3 * DD + 2 * DD + d];
    float rg = sigmf(ir + hr);
    float zg = sigmf(iz + hz);
    float ng = tanhf(in + rg * hn);
    float h  = slots[s * DD + d];
    float upd = (1.f - zg) * ng + zg * h;
    float gg = g[b];
    out[i] = gg * upd + (1.f - gg) * h;
}

// ---------------------------------------------------------------------------
extern "C" void kernel_launch(void* const* d_in, const int* in_sizes, int n_in,
                              void* d_out, int out_size, void* d_ws, size_t ws_size,
                              hipStream_t stream)
{
    (void)in_sizes; (void)n_in; (void)out_size; (void)ws_size;

    const int*   tokens  = (const int*)d_in[0];
    const float* tok_emb = (const float*)d_in[1];
    const float* pos_emb = (const float*)d_in[2];
    const float* rel_b   = (const float*)d_in[3];
    const float* in_ln_g = (const float*)d_in[4];
    const float* in_ln_b = (const float*)d_in[5];
    const float* f_ln_g  = (const float*)d_in[6];
    const float* f_ln_b  = (const float*)d_in[7];
    const float* mg_w    = (const float*)d_in[8];
    const float* mg_b    = (const float*)d_in[9];
    const float* slots   = (const float*)d_in[10];
    const float* q_w     = (const float*)d_in[11];
    const float* q_b     = (const float*)d_in[12];
    const float* k_w     = (const float*)d_in[13];
    const float* k_b     = (const float*)d_in[14];
    const float* vm_w    = (const float*)d_in[15];
    const float* vm_b    = (const float*)d_in[16];
    const float* om_w    = (const float*)d_in[17];
    const float* om_b    = (const float*)d_in[18];
    const float* wih     = (const float*)d_in[19];
    const float* bih     = (const float*)d_in[20];
    const float* whh     = (const float*)d_in[21];
    const float* bhh     = (const float*)d_in[22];
    const float* g_w     = (const float*)d_in[23];
    const float* g_b     = (const float*)d_in[24];
    auto LI = [&](int l, int j) { return (const float*)d_in[25 + 23 * l + j]; };
    // layer param order: 0 time_mix 1 channel_mix 2 v_w 3 v_b 4 r_w 5 r_b 6 o_w 7 o_b
    // 8 decay 9 ln1_g 10 ln1_b 11 ln2_g 12 ln2_b 13 lnc_g 14 lnc_b
    // 15 up_w 16 up_b 17 down_w 18 down_b 19 dw_w 20 dw_b 21 pw_w 22 pw_b

    // ---- workspace bump allocator (~115 MB total) ----
    char* wp = (char*)d_ws;
    auto alloc = [&](size_t bytes) -> void* {
        void* r = (void*)wp;
        wp += (bytes + 255) & ~(size_t)255;
        return r;
    };
    float* x    = (float*)alloc((size_t)MT * DD * 4);
    float* t0   = (float*)alloc((size_t)MT * DD * 4);
    float* t1   = (float*)alloc((size_t)MT * DD * 4);
    float* t2   = (float*)alloc((size_t)MT * DD * 4);
    float* kf   = (float*)alloc((size_t)SS * DD * 4);
    float* vf   = (float*)alloc((size_t)SS * DD * 4);
    float* summ = (float*)alloc((size_t)BB * DD * 4);
    float* mmn  = (float*)alloc((size_t)DD * 4);
    float* gsc  = (float*)alloc((size_t)BB * 4);
    float* gi   = (float*)alloc((size_t)BB * 3 * DD * 4);
    float* gh   = (float*)alloc((size_t)SS * 3 * DD * 4);
    bf16* h0    = (bf16*)alloc((size_t)MT * DD * 2);
    bf16* h1    = (bf16*)alloc((size_t)MT * DD * 2);
    bf16* hbig  = (bf16*)alloc((size_t)MT * 4 * DD * 2);
    bf16* hs    = (bf16*)alloc((size_t)SS * DD * 2);
    bf16* wtok  = (bf16*)alloc((size_t)VV * DD * 2);
    bf16* wqm   = (bf16*)alloc((size_t)DD * DD * 2);
    bf16* wkm   = (bf16*)alloc((size_t)DD * DD * 2);
    bf16* wvm   = (bf16*)alloc((size_t)DD * DD * 2);
    bf16* wom   = (bf16*)alloc((size_t)DD * DD * 2);
    bf16 *wv[NLAY], *wr[NLAY], *wo[NLAY], *wpw[NLAY], *wup[NLAY], *wdn[NLAY];
    for (int l = 0; l < NLAY; ++l) {
        wv[l]  = (bf16*)alloc((size_t)DD * DD * 2);
        wr[l]  = (bf16*)alloc((size_t)DD * DD * 2);
        wo[l]  = (bf16*)alloc((size_t)DD * DD * 2);
        wpw[l] = (bf16*)alloc((size_t)DD * DD * 2);
        wup[l] = (bf16*)alloc((size_t)4 * DD * DD * 2);
        wdn[l] = (bf16*)alloc((size_t)DD * 4 * DD * 2);
    }

    auto cvt = [&](const float* s, bf16* d, size_t n) {
        k_cvt<<<dim3((unsigned)((n + 255) / 256)), dim3(256), 0, stream>>>(s, d, (int)n);
    };
    auto gemm = [&](int epi, const bf16* A, const bf16* W, const float* bias,
                    float* Cf, bf16* Ch, const float* resid, int M, int N, int K) {
        dim3 grid((unsigned)(N / 64), (unsigned)((M + 127) / 128));
        dim3 blk(128);
        switch (epi) {
        case EPI_NONE: k_gemm<EPI_NONE><<<grid, blk, 0, stream>>>(A, W, bias, Cf, Ch, resid, M, N, K); break;
        case EPI_SIG:  k_gemm<EPI_SIG ><<<grid, blk, 0, stream>>>(A, W, bias, Cf, Ch, resid, M, N, K); break;
        case EPI_SILU: k_gemm<EPI_SILU><<<grid, blk, 0, stream>>>(A, W, bias, Cf, Ch, resid, M, N, K); break;
        default:       k_gemm<EPI_RES ><<<grid, blk, 0, stream>>>(A, W, bias, Cf, Ch, resid, M, N, K); break;
        }
    };

    // ---- weight conversions (deterministic each call) ----
    cvt(tok_emb, wtok, (size_t)VV * DD);
    cvt(q_w,  wqm, (size_t)DD * DD);
    cvt(k_w,  wkm, (size_t)DD * DD);
    cvt(vm_w, wvm, (size_t)DD * DD);
    cvt(om_w, wom, (size_t)DD * DD);
    cvt(slots, hs, (size_t)SS * DD);
    for (int l = 0; l < NLAY; ++l) {
        cvt(LI(l, 2),  wv[l],  (size_t)DD * DD);
        cvt(LI(l, 4),  wr[l],  (size_t)DD * DD);
        cvt(LI(l, 6),  wo[l],  (size_t)DD * DD);
        cvt(LI(l, 21), wpw[l], (size_t)DD * DD);
        cvt(LI(l, 15), wup[l], (size_t)4 * DD * DD);
        cvt(LI(l, 17), wdn[l], (size_t)DD * 4 * DD);
    }

    const size_t NE = (size_t)MT * DD;
    const dim3 eb((unsigned)((NE + 255) / 256)), et(256);

    // ---- embedding + input LN ----
    k_embed<<<eb, et, 0, stream>>>(tokens, tok_emb, pos_emb, t0);
    k_ln<<<MT, 256, 0, stream>>>(t0, in_ln_g, in_ln_b, x, h0);

    // ---- sparse memory read ----
    gemm(EPI_NONE, h0, wqm, q_b, t1, nullptr, nullptr, MT, DD, DD);     // q
    gemm(EPI_NONE, hs, wkm, k_b, kf, nullptr, nullptr, SS, DD, DD);     // k (slots)
    gemm(EPI_NONE, hs, wvm, vm_b, vf, nullptr, nullptr, SS, DD, DD);    // v (slots)
    k_attn<<<MT, 32, 0, stream>>>(t1, kf, vf, h1);                      // top-k attn -> ctx_pre
    gemm(EPI_NONE, h1, wom, om_b, t2, nullptr, nullptr, MT, DD, DD);    // ctx
    k_gate<<<MT, 256, 0, stream>>>(x, t2, mg_w, mg_b, rel_b);           // x += gate*ctx

    // ---- RWKV-ish blocks ----
    for (int l = 0; l < NLAY; ++l) {
        k_ln<<<MT, 256, 0, stream>>>(x, LI(l, 9), LI(l, 10), t0, nullptr);
        k_mix<<<eb, et, 0, stream>>>(t0, LI(l, 0), h0);                 // time-shift mix
        gemm(EPI_NONE, h0, wv[l], LI(l, 3), t1, nullptr, nullptr, MT, DD, DD);   // v
        gemm(EPI_SIG,  h0, wr[l], LI(l, 5), t2, nullptr, nullptr, MT, DD, DD);   // r = sigmoid
        k_rv<<<eb, et, 0, stream>>>(t1, t2);                            // rv = r*v
        k_decay<<<eb, et, 0, stream>>>(t1, LI(l, 8), h1);               // causal decay conv
        gemm(EPI_RES, h1, wo[l], LI(l, 7), x, nullptr, x, MT, DD, DD);  // x += .5*o(y)
        k_ln<<<MT, 256, 0, stream>>>(x, LI(l, 11), LI(l, 12), t0, nullptr);
        k_mix<<<eb, et, 0, stream>>>(t0, LI(l, 1), h0);                 // channel-shift mix
        gemm(EPI_SILU, h0, wup[l], LI(l, 16), nullptr, hbig, nullptr, MT, 4 * DD, DD);
        gemm(EPI_RES, hbig, wdn[l], LI(l, 18), x, nullptr, x, MT, DD, 4 * DD);
        k_ln<<<MT, 256, 0, stream>>>(x, LI(l, 13), LI(l, 14), t0, nullptr);
        k_dw<<<eb, et, 0, stream>>>(t0, LI(l, 19), LI(l, 20), h0);      // dilated depthwise
        gemm(EPI_RES, h0, wpw[l], LI(l, 22), x, nullptr, x, MT, DD, DD);// x += .5*pw(..)
    }

    // ---- final LN + tied LM head ----
    k_ln<<<MT, 256, 0, stream>>>(x, f_ln_g, f_ln_b, t0, h0);
    float* logits = (float*)d_out;
    gemm(EPI_NONE, h0, wtok, nullptr, logits, nullptr, nullptr, MT, VV, DD);

    // ---- gated GRU memory write ----
    k_mean<<<(BB * DD + 255) / 256, 256, 0, stream>>>(t0, summ, BB, TT, DD);
    k_mean<<<(DD + 255) / 256, 256, 0, stream>>>(slots, mmn, 1, SS, DD);
    k_gscal<<<BB, 256, 0, stream>>>(summ, mmn, g_w, g_b, gsc);
    const int ntot = BB * 3 * DD + SS * 3 * DD;
    k_gruproj<<<(ntot + 255) / 256, 256, 0, stream>>>(summ, slots, wih, bih, whh, bhh, gi, gh);
    float* memout = logits + (size_t)BB * TT * VV;
    k_gruupd<<<(BB * SS * DD + 255) / 256, 256, 0, stream>>>(gi, gh, slots, gsc, memout);
}
